// MultiHeadAttention_12996571037820
// MI455X (gfx1250) — matrix operand order repaired and checked by reading, not compiled, not run on hardware
//
#include <hip/hip_runtime.h>
#include <hip/hip_bf16.h>
#include <stdint.h>

typedef __bf16 bf16;
typedef __attribute__((ext_vector_type(16))) __bf16 v16bf;
typedef __attribute__((ext_vector_type(8)))  __bf16 v8bf;
typedef __attribute__((ext_vector_type(4)))  __bf16 v4bf;
typedef __attribute__((ext_vector_type(8)))  float  v8f;
typedef __attribute__((ext_vector_type(4)))  float  v4f;

#define D_MODEL 512
#define DK      64
#define NH      8
#define SEQ     4096
#define BATCH   2
#define MTOT    (BATCH * SEQ)   // 8192

union ABFrag { v16bf v; v8bf h[2]; };

__device__ __forceinline__ v8f wmma_bf16f32(v16bf a, v16bf b, v8f c) {
    // (neg_a, A, neg_b, B, c_mod, C, reuse_a, reuse_b)
    return __builtin_amdgcn_wmma_f32_16x16x32_bf16(false, a, false, b, (short)0, c,
                                                   false, false);
}

// A fragment: 16x32 tile of a row-major [*, ld] bf16 matrix.
// lane<16: row=lane,    elems = {k0+0..7, k0+16..23}
// lane>=16: row=lane-16, elems = {k0+8..15, k0+24..31}
__device__ __forceinline__ v16bf load_a_frag(const bf16* base, int ld, int row0,
                                             int k0, int lane) {
    int r = lane & 15, half = lane >> 4;
    const bf16* p = base + (size_t)(row0 + r) * ld + k0 + half * 8;
    ABFrag f;
    f.h[0] = *(const v8bf*)(p);
    f.h[1] = *(const v8bf*)(p + 16);
    return f.v;
}

// B fragment: 32x16 tile (contraction x N); source stored row-major [N, ld]
// with row index = output column n.
// lane<16: n=lane,    elems {k0+0..7, k0+16..23} ; lane>=16: n=lane-16, elems {k0+8..15, k0+24..31}
__device__ __forceinline__ v16bf load_b_frag(const bf16* base, int ld, int n0,
                                             int k0, int lane) {
    int n = lane & 15, half = lane >> 4;
    const bf16* p = base + (size_t)(n0 + n) * ld + k0 + half * 8;
    ABFrag f;
    f.h[0] = *(const v8bf*)(p);
    f.h[1] = *(const v8bf*)(p + 16);
    return f.v;
}

// ---- CDNA5 async global->LDS (ASYNCcnt) helpers ----
// Low 32 bits of a generic LDS address are the LDS byte offset (ISA 10.2).
__device__ __forceinline__ uint32_t lds_off(const void* p) {
    return (uint32_t)(uintptr_t)p;
}
__device__ __forceinline__ void async_load_b128(uint32_t lds_byte_off, const bf16* gptr) {
    asm volatile("global_load_async_to_lds_b128 %0, %1, off"
                 :: "v"(lds_byte_off), "v"(gptr) : "memory");
}
__device__ __forceinline__ void wait_asynccnt0() {
    asm volatile("s_wait_asynccnt 0" ::: "memory");
}

// ---------------- f32 -> bf16 convert (x4 vectorized) ----------------
__global__ __launch_bounds__(256)
void cvt_f32_bf16_x4(const float* __restrict__ src, bf16* __restrict__ dst, int n4) {
    int i = blockIdx.x * 256 + threadIdx.x;
    if (i < n4) {
        v4f v = *(const v4f*)(src + (size_t)i * 4);
        v4bf o;
        o[0] = (bf16)v[0]; o[1] = (bf16)v[1]; o[2] = (bf16)v[2]; o[3] = (bf16)v[3];
        *(v4bf*)(dst + (size_t)i * 4) = o;
    }
}

// ---------------- WMMA GEMM: C = A[M,K] * W[N,K]^T + bias ----------------
enum { MODE_QK = 0, MODE_VT = 1, MODE_OUT = 2 };

template <int MODE>
__global__ __launch_bounds__(256)
void gemm_wmma(const bf16* __restrict__ A, const bf16* __restrict__ W,
               const float* __restrict__ bias, void* __restrict__ Out,
               int M, int N, int K) {
    int lane = threadIdx.x & 31;
    int wave = threadIdx.x >> 5;
    int r15 = lane & 15, half = lane >> 4;
    int m0 = blockIdx.y * 128 + wave * 16;   // block tile: 128(M) x 64(N), 8 waves
    int n0 = blockIdx.x * 64;

    v8f acc[4];
#pragma unroll
    for (int t = 0; t < 4; ++t)
#pragma unroll
        for (int r = 0; r < 8; ++r) acc[t][r] = 0.0f;

    for (int k0 = 0; k0 < K; k0 += 32) {
        v16bf af = load_a_frag(A, K, m0, k0, lane);
#pragma unroll
        for (int t = 0; t < 4; ++t) {
            v16bf wf = load_b_frag(W, K, n0 + t * 16, k0, lane);
            acc[t] = wmma_bf16f32(af, wf, acc[t]);
        }
    }

#pragma unroll
    for (int t = 0; t < 4; ++t) {
        float bv = bias[n0 + t * 16 + r15];
#pragma unroll
        for (int r = 0; r < 8; ++r) acc[t][r] += bv;
    }

    if (MODE == MODE_OUT) {
        float* out = (float*)Out;
#pragma unroll
        for (int t = 0; t < 4; ++t) {
            int n = n0 + t * 16 + r15;
#pragma unroll
            for (int r = 0; r < 8; ++r) {
                int m = m0 + half * 8 + r;
                out[(size_t)m * N + n] = acc[t][r];
            }
        }
    } else if (MODE == MODE_QK) {
        // store bf16 as [B*H, S, DK]
        bf16* out = (bf16*)Out;
#pragma unroll
        for (int t = 0; t < 4; ++t) {
            int n = n0 + t * 16 + r15;
            int hh = n >> 6, dk = n & 63;
#pragma unroll
            for (int r = 0; r < 8; ++r) {
                int m = m0 + half * 8 + r;
                int bb = m >> 12, s = m & (SEQ - 1);
                out[((size_t)(bb * NH + hh) * SEQ + s) * DK + dk] = (bf16)acc[t][r];
            }
        }
    } else {
        // MODE_VT: store bf16 transposed as [B*H, DK, S] (contiguous 16B per lane)
        bf16* out = (bf16*)Out;
        int bb = m0 >> 12;
        int s0 = (m0 & (SEQ - 1)) + half * 8;
#pragma unroll
        for (int t = 0; t < 4; ++t) {
            int n = n0 + t * 16 + r15;
            int hh = n >> 6, dk = n & 63;
            v8bf pk;
#pragma unroll
            for (int r = 0; r < 8; ++r) pk[r] = (bf16)acc[t][r];
            *(v8bf*)(out + ((size_t)(bb * NH + hh) * DK + dk) * SEQ + s0) = pk;
        }
    }
}

// Stage one 64-key K tile (64x64 bf16) and V tile (64dk x 64key bf16) into LDS
// via CDNA5 async global->LDS DMA. 256 threads, 4 x b128 issues per thread.
__device__ __forceinline__ void stage_kv_async(uint32_t kbase, uint32_t vbase,
                                               const bf16* Kb, const bf16* Vb,
                                               int kb, int tid) {
#pragma unroll
    for (int j = 0; j < 2; ++j) {
        int c = tid * 2 + j;            // 0..511 16-byte chunks
        int row = c >> 3;               // 0..63
        int col = (c & 7) * 8;          // element offset, 16B granules
        async_load_b128(kbase + (uint32_t)c * 16, Kb + (size_t)(kb + row) * DK + col);
        async_load_b128(vbase + (uint32_t)c * 16, Vb + (size_t)row * SEQ + kb + col);
    }
}

// ---------------- Flash attention: one wave = 16 query rows ----------------
__global__ __launch_bounds__(256)
void flash_attn(const bf16* __restrict__ Q,   // [B*H, S, DK]
                const bf16* __restrict__ Kh,  // [B*H, S, DK]
                const bf16* __restrict__ Vt,  // [B*H, DK, S]
                bf16* __restrict__ ctx) {     // [B, S, D_MODEL] (d = h*64+dk)
    __shared__ __align__(16) bf16 Ktile[2][64 * 64];  // [key-kb][dk], 8KB each
    __shared__ __align__(16) bf16 Vtile[2][64 * 64];  // [dk][key-kb], 8KB each
    __shared__ __align__(16) bf16 Pbuf[8][16 * 64];   // per-wave P staging

    int tid = threadIdx.x;
    int lane = tid & 31, wave = tid >> 5;
    int r15 = lane & 15, half = lane >> 4;
    int bh = blockIdx.x >> 5;                 // 16 (b,h) pairs
    int qtile = blockIdx.x & 31;              // 32 query tiles of 128
    int b = bh >> 3, h = bh & 7;
    int q0 = qtile * 128 + wave * 16;

    const bf16* Qb = Q  + (size_t)bh * SEQ * DK;
    const bf16* Kb = Kh + (size_t)bh * SEQ * DK;
    const bf16* Vb = Vt + (size_t)bh * DK * SEQ;

    uint32_t koff[2] = { lds_off(&Ktile[0][0]), lds_off(&Ktile[1][0]) };
    uint32_t voff[2] = { lds_off(&Vtile[0][0]), lds_off(&Vtile[1][0]) };

    v16bf qf0 = load_a_frag(Qb, DK, q0, 0, lane);
    v16bf qf1 = load_a_frag(Qb, DK, q0, 32, lane);

    v8f O[4];
    float m_run[8], l_run[8];
#pragma unroll
    for (int t = 0; t < 4; ++t)
#pragma unroll
        for (int r = 0; r < 8; ++r) O[t][r] = 0.0f;
#pragma unroll
    for (int r = 0; r < 8; ++r) { m_run[r] = -1.0e30f; l_run[r] = 0.0f; }

    // work in exp2 domain: sl = s * (1/sqrt(dk)) * log2(e)
    const float scale_log2e = 0.125f * 1.44269504088896f;

    // prologue: start DMA of first key block
    stage_kv_async(koff[0], voff[0], Kb, Vb, 0, tid);

    for (int kb = 0; kb < SEQ; kb += 64) {
        int cur = (kb >> 6) & 1;
        wait_asynccnt0();       // my tile-DMA for buf[cur] done
        __syncthreads();        // everyone's done; prior reads of buf[cur^1] done
        if (kb + 64 < SEQ)
            stage_kv_async(koff[cur ^ 1], voff[cur ^ 1], Kb, Vb, kb + 64, tid);

        const bf16* Kt  = &Ktile[cur][0];
        const bf16* Vtl = &Vtile[cur][0];

        v8f sc[4];
#pragma unroll
        for (int t = 0; t < 4; ++t) {
            v16bf kf0 = load_b_frag(Kt, 64, t * 16, 0, lane);
            v16bf kf1 = load_b_frag(Kt, 64, t * 16, 32, lane);
            v8f s;
#pragma unroll
            for (int r = 0; r < 8; ++r) s[r] = 0.0f;
            s = wmma_bf16f32(qf0, kf0, s);
            s = wmma_bf16f32(qf1, kf1, s);
#pragma unroll
            for (int r = 0; r < 8; ++r) sc[t][r] = s[r] * scale_log2e;
        }

        // per-row max across 64 keys (4 tiles x 16 lanes within half-wave)
        float mx[8];
#pragma unroll
        for (int r = 0; r < 8; ++r)
            mx[r] = fmaxf(fmaxf(sc[0][r], sc[1][r]), fmaxf(sc[2][r], sc[3][r]));
#pragma unroll
        for (int d = 8; d >= 1; d >>= 1)
#pragma unroll
            for (int r = 0; r < 8; ++r)
                mx[r] = fmaxf(mx[r], __shfl_xor(mx[r], d, 32));

        float fac[8], rs[8];
#pragma unroll
        for (int r = 0; r < 8; ++r) {
            float mn = fmaxf(m_run[r], mx[r]);
            fac[r] = exp2f(m_run[r] - mn);
            m_run[r] = mn;
            rs[r] = 0.0f;
        }

        // p = exp2(sl - m); write to LDS in (q, key) layout for the A-fragment
#pragma unroll
        for (int t = 0; t < 4; ++t)
#pragma unroll
            for (int r = 0; r < 8; ++r) {
                float p = exp2f(sc[t][r] - m_run[r]);
                rs[r] += p;
                Pbuf[wave][(half * 8 + r) * 64 + t * 16 + r15] = (bf16)p;
            }
#pragma unroll
        for (int d = 8; d >= 1; d >>= 1)
#pragma unroll
            for (int r = 0; r < 8; ++r) rs[r] += __shfl_xor(rs[r], d, 32);
#pragma unroll
        for (int r = 0; r < 8; ++r) l_run[r] = l_run[r] * fac[r] + rs[r];

        // rescale running output
#pragma unroll
        for (int t = 0; t < 4; ++t)
#pragma unroll
            for (int r = 0; r < 8; ++r) O[t][r] *= fac[r];

        // O += P[16x64] * V[64x64]  (same-wave DS ops are in-order, no barrier)
#pragma unroll
        for (int ks = 0; ks < 64; ks += 32) {
            v16bf pf = load_a_frag(&Pbuf[wave][0], 64, 0, ks, lane);
#pragma unroll
            for (int tn = 0; tn < 4; ++tn) {
                v16bf vf = load_b_frag(Vtl, 64, tn * 16, ks, lane);
                O[tn] = wmma_bf16f32(pf, vf, O[tn]);
            }
        }
    }

    // epilogue: normalize and scatter to ctx[b, s, h*64+dk] (bf16)
#pragma unroll
    for (int r = 0; r < 8; ++r) {
        float inv = 1.0f / l_run[r];
        int srow = q0 + half * 8 + r;
        size_t rowoff = ((size_t)b * SEQ + srow) * D_MODEL + h * DK;
#pragma unroll
        for (int t = 0; t < 4; ++t)
            ctx[rowoff + t * 16 + r15] = (bf16)(O[t][r] * inv);
    }
}

extern "C" void kernel_launch(void* const* d_in, const int* in_sizes, int n_in,
                              void* d_out, int out_size, void* d_ws, size_t ws_size,
                              hipStream_t stream) {
    (void)in_sizes; (void)n_in; (void)out_size; (void)ws_size;
    const float* x  = (const float*)d_in[0];
    const float* Wq = (const float*)d_in[1];
    const float* bq = (const float*)d_in[2];
    const float* Wk = (const float*)d_in[3];
    const float* bk = (const float*)d_in[4];
    const float* Wv = (const float*)d_in[5];
    const float* bv = (const float*)d_in[6];
    const float* Wo = (const float*)d_in[7];
    const float* bo = (const float*)d_in[8];

    size_t off = 0;
    auto alloc = [&](size_t bytes) -> void* {
        void* p = (char*)d_ws + off;
        off += (bytes + 255) & ~(size_t)255;
        return p;
    };
    const size_t nX = (size_t)MTOT * D_MODEL;      // 4,194,304
    const size_t nW = (size_t)D_MODEL * D_MODEL;   // 262,144

    bf16* xb  = (bf16*)alloc(nX * 2);
    bf16* wqb = (bf16*)alloc(nW * 2);
    bf16* wkb = (bf16*)alloc(nW * 2);
    bf16* wvb = (bf16*)alloc(nW * 2);
    bf16* wob = (bf16*)alloc(nW * 2);
    bf16* Qb  = (bf16*)alloc(nX * 2);
    bf16* Kb  = (bf16*)alloc(nX * 2);
    bf16* Vtb = (bf16*)alloc(nX * 2);
    bf16* ctx = (bf16*)alloc(nX * 2);

    cvt_f32_bf16_x4<<<(int)(nX / 1024), 256, 0, stream>>>(x,  xb,  (int)(nX / 4));
    cvt_f32_bf16_x4<<<(int)(nW / 1024), 256, 0, stream>>>(Wq, wqb, (int)(nW / 4));
    cvt_f32_bf16_x4<<<(int)(nW / 1024), 256, 0, stream>>>(Wk, wkb, (int)(nW / 4));
    cvt_f32_bf16_x4<<<(int)(nW / 1024), 256, 0, stream>>>(Wv, wvb, (int)(nW / 4));
    cvt_f32_bf16_x4<<<(int)(nW / 1024), 256, 0, stream>>>(Wo, wob, (int)(nW / 4));

    dim3 ggrid(D_MODEL / 64, MTOT / 128);  // (8, 64)
    gemm_wmma<MODE_QK ><<<ggrid, 256, 0, stream>>>(xb, wqb, bq, Qb,  MTOT, D_MODEL, D_MODEL);
    gemm_wmma<MODE_QK ><<<ggrid, 256, 0, stream>>>(xb, wkb, bk, Kb,  MTOT, D_MODEL, D_MODEL);
    gemm_wmma<MODE_VT ><<<ggrid, 256, 0, stream>>>(xb, wvb, bv, Vtb, MTOT, D_MODEL, D_MODEL);

    flash_attn<<<BATCH * NH * (SEQ / 128), 256, 0, stream>>>(Qb, Kb, Vtb, ctx);

    gemm_wmma<MODE_OUT><<<ggrid, 256, 0, stream>>>(ctx, wob, bo, d_out, MTOT, D_MODEL, D_MODEL);
}
